// AttentionModule_69698729279969
// MI455X (gfx1250) — compile-verified, matrix operand order checked by reading, not run on hardware
//
#include <hip/hip_runtime.h>
#include <hip/hip_bf16.h>

typedef __attribute__((ext_vector_type(8)))  float   v8f;
typedef __attribute__((ext_vector_type(8)))  __bf16  v8bf;
typedef __attribute__((ext_vector_type(16))) __bf16  v16bf;

union TileVec { v16bf full; struct { v8bf lo, hi; } h; };

#define B_SZ   2
#define S_LEN  2048
#define D_DIM  1024
#define H_NUM  16
#define HD     64
#define M_TOT  (B_SZ * S_LEN)   // 4096
#define SCALE_F 0.125f          // 1/sqrt(64)

__device__ __forceinline__ v8f wmma_bf16(v16bf a, v16bf b, v8f c) {
    return __builtin_amdgcn_wmma_f32_16x16x32_bf16(
        false, a, false, b, (short)0, c, false, false);
}

// Global 16x32 (16-bit) WMMA A/B operand tile (both use the same lane pattern):
//   lane l -> row (row0 + (l&15)), two 8-elem chunks at kc+(l>>4)*8 and +16.
__device__ __forceinline__ v16bf load_tile16(const __bf16* base, int row0,
                                             int rowStride, int kc) {
    const int l = threadIdx.x & 31;
    const __bf16* p = base + (size_t)(row0 + (l & 15)) * rowStride
                           + kc + ((l >> 4) << 3);
    TileVec t;
    t.h.lo = *(const v8bf*)p;
    t.h.hi = *(const v8bf*)(p + 16);
    return t.full;
}

// Same tile pattern read out of an LDS panel laid out row-major [rows][32].
__device__ __forceinline__ v16bf lds_tile16(const __bf16 (*buf)[32], int row0) {
    const int l = threadIdx.x & 31;
    const __bf16* p = &buf[row0 + (l & 15)][(l >> 4) << 3];
    TileVec t;
    t.h.lo = *(const v8bf*)p;
    t.h.hi = *(const v8bf*)(p + 16);
    return t.full;
}

// Async global -> LDS, 16B per lane, tracked by ASYNCcnt.
__device__ __forceinline__ void async_b128(const __bf16* gaddr, const __bf16* laddr) {
    const unsigned lds_off = (unsigned)(unsigned long long)laddr; // low 32b = LDS offset
    asm volatile("global_load_async_to_lds_b128 %0, %1, off"
                 :: "v"(lds_off), "v"(gaddr) : "memory");
}
__device__ __forceinline__ void wait_async0() {
    asm volatile("s_wait_asynccnt 0x0" ::: "memory");
}

// ---------------------------------------------------------------- convert ---
__global__ void cvt_f32_bf16(const float* __restrict__ src,
                             __bf16* __restrict__ dst, int n) {
    int i = blockIdx.x * blockDim.x + threadIdx.x;
    if (i < n) dst[i] = (__bf16)src[i];
}

// ------------------------------------------------------------------- GEMM ---
// y[m,n] = sum_k A[m,k] * W[n,k] + bias[n]
// Block: 256 thr = 8 waves (2M x 4N), block tile 128x128, wave tile 64x32.
// A/W panels (128x32 bf16) staged per k-step via async global->LDS loads,
// double-buffered; branchless wrap-around prefetch; k-loop unrolled by 2 so
// the buffer index is compile-time constant.
// mode 0: bf16 [M_TOT,1024]   mode 1: bf16 V-transposed   mode 2: fp32
__global__ __launch_bounds__(256)
void gemm_xw(const __bf16* __restrict__ A, const __bf16* __restrict__ W,
             const float* __restrict__ bias, void* __restrict__ outp, int mode) {
    __shared__ __align__(16) __bf16 smA[2][128][32];   // 2 x 8 KiB
    __shared__ __align__(16) __bf16 smW[2][128][32];   // 2 x 8 KiB

    const int lane = threadIdx.x & 31;
    const int wave = threadIdx.x >> 5;
    const int Mblk = blockIdx.x * 128;
    const int Nblk = blockIdx.y * 128;
    const int Mwave = (wave >> 2) * 64;
    const int Nwave = (wave & 3) * 32;
    const int half = lane >> 4, nlo = lane & 15;

    // Per-thread staging slots: 2 chunks of A and 2 of W per k-step.
    const int ch0 = threadIdx.x,       row0 = ch0 >> 2, part0 = (ch0 & 3) << 3;
    const int ch1 = threadIdx.x + 256, row1 = ch1 >> 2, part1 = (ch1 & 3) << 3;
    const __bf16* gA0 = A + (size_t)(Mblk + row0) * D_DIM + part0;
    const __bf16* gA1 = A + (size_t)(Mblk + row1) * D_DIM + part1;
    const __bf16* gW0 = W + (size_t)(Nblk + row0) * D_DIM + part0;
    const __bf16* gW1 = W + (size_t)(Nblk + row1) * D_DIM + part1;

    auto issue = [&](int kc, int bufIdx) {
        async_b128(gA0 + kc, &smA[bufIdx][row0][part0]);
        async_b128(gA1 + kc, &smA[bufIdx][row1][part1]);
        async_b128(gW0 + kc, &smW[bufIdx][row0][part0]);
        async_b128(gW1 + kc, &smW[bufIdx][row1][part1]);
    };

    v8f acc[4][2] = {};
    constexpr int KSTEPS = D_DIM / 32;   // 32 (even)

    auto stage = [&](int kNext, int bufCompute) {
        wait_async0();      // my writes for the current stage are in LDS
        __syncthreads();    // everyone's writes done; prev buffer fully consumed
        issue((kNext & (KSTEPS - 1)) * 32, bufCompute ^ 1);  // branchless wrap
        v16bf b0 = lds_tile16(smW[bufCompute], Nwave);
        v16bf b1 = lds_tile16(smW[bufCompute], Nwave + 16);
#pragma unroll
        for (int mt = 0; mt < 4; ++mt) {
            v16bf a = lds_tile16(smA[bufCompute], Mwave + mt * 16);
            acc[mt][0] = wmma_bf16(a, b0, acc[mt][0]);
            acc[mt][1] = wmma_bf16(a, b1, acc[mt][1]);
        }
    };

    issue(0, 0);
    for (int k = 0; k < KSTEPS; k += 2) {
        stage(k + 1, 0);    // compute buffer 0, prefetch k+1 into buffer 1
        stage(k + 2, 1);    // compute buffer 1, prefetch k+2 into buffer 0
    }
    wait_async0();          // drain the harmless wrap-around prefetch

    const int Nbase = Nblk + Nwave;
    const int Mbase = Mblk + Mwave;
    const float bAdd[2] = { bias[Nbase + nlo], bias[Nbase + 16 + nlo] };

    if (mode == 0) {
        __bf16* out = (__bf16*)outp;
#pragma unroll
        for (int mt = 0; mt < 4; ++mt)
#pragma unroll
            for (int nt = 0; nt < 2; ++nt) {
                const int n = Nbase + nt * 16 + nlo;
#pragma unroll
                for (int v = 0; v < 8; ++v) {
                    const int m = Mbase + mt * 16 + 8 * half + v;
                    out[(size_t)m * D_DIM + n] = (__bf16)(acc[mt][nt][v] + bAdd[nt]);
                }
            }
    } else if (mode == 1) {
        __bf16* vt = (__bf16*)outp;
#pragma unroll
        for (int mt = 0; mt < 4; ++mt)
#pragma unroll
            for (int nt = 0; nt < 2; ++nt) {
                const int n  = Nbase + nt * 16 + nlo;
                const int hh = n >> 6, hd = n & 63;
                const int mrow = Mbase + mt * 16;
                const int bb = mrow >> 11;
                const int s0 = (mrow & (S_LEN - 1)) + 8 * half;
                v8bf pk;
#pragma unroll
                for (int v = 0; v < 8; ++v)
                    pk[v] = (__bf16)(acc[mt][nt][v] + bAdd[nt]);
                *(v8bf*)(vt + ((size_t)(bb * H_NUM + hh) * HD + hd) * S_LEN + s0) = pk;
            }
    } else {
        float* out = (float*)outp;
#pragma unroll
        for (int mt = 0; mt < 4; ++mt)
#pragma unroll
            for (int nt = 0; nt < 2; ++nt) {
                const int n = Nbase + nt * 16 + nlo;
#pragma unroll
                for (int v = 0; v < 8; ++v) {
                    const int m = Mbase + mt * 16 + 8 * half + v;
                    out[(size_t)m * D_DIM + n] = acc[mt][nt][v] + bAdd[nt];
                }
            }
    }
}

// -------------------------------------------------------- flash attention ---
// One wave per (b, h, 16-row q-tile). Online softmax over k-chunks of 32.
__global__ __launch_bounds__(256)
void attn_flash(const __bf16* __restrict__ qb, const __bf16* __restrict__ kb,
                const __bf16* __restrict__ vt, const float* __restrict__ mask,
                __bf16* __restrict__ ob) {
    __shared__ __align__(16) __bf16 lds[8][16 * 32];   // per-wave P staging

    const int lane = threadIdx.x & 31;
    const int wave = threadIdx.x >> 5;
    const int gw = blockIdx.x * 8 + wave;
    const int qt = gw & 127;
    const int h  = (gw >> 7) & (H_NUM - 1);
    const int b  = gw >> 11;
    const int q0 = qt * 16;
    const int half = lane >> 4, nlo = lane & 15;

    const __bf16* qbase = qb + ((size_t)b * S_LEN) * D_DIM + (size_t)h * HD;
    const __bf16* kbase = kb + ((size_t)b * S_LEN) * D_DIM + (size_t)h * HD;
    const __bf16* vbase = vt + (size_t)(b * H_NUM + h) * HD * S_LEN;
    const float*  mbase = mask + (size_t)b * S_LEN * S_LEN;

    const v16bf qa0 = load_tile16(qbase, q0, D_DIM, 0);
    const v16bf qa1 = load_tile16(qbase, q0, D_DIM, 32);

    v8f o[4] = {};
    float mrow[8], lrow[8];
#pragma unroll
    for (int v = 0; v < 8; ++v) { mrow[v] = -1e30f; lrow[v] = 0.0f; }

    for (int kc0 = 0; kc0 < S_LEN; kc0 += 32) {
        v8f s[2];
#pragma unroll
        for (int t = 0; t < 2; ++t) {
            v16bf kt0 = load_tile16(kbase, kc0 + t * 16, D_DIM, 0);
            v16bf kt1 = load_tile16(kbase, kc0 + t * 16, D_DIM, 32);
            v8f c = {};
            c = wmma_bf16(qa0, kt0, c);
            c = wmma_bf16(qa1, kt1, c);
#pragma unroll
            for (int v = 0; v < 8; ++v) {
                const int qr  = q0 + 8 * half + v;
                const int col = kc0 + t * 16 + nlo;
                c[v] = c[v] * SCALE_F + mbase[(size_t)qr * S_LEN + col];
            }
            s[t] = c;
        }
        float pscale[8];
#pragma unroll
        for (int v = 0; v < 8; ++v) {
            float mx = fmaxf(s[0][v], s[1][v]);
            mx = fmaxf(mx, __shfl_xor(mx, 1, 32));
            mx = fmaxf(mx, __shfl_xor(mx, 2, 32));
            mx = fmaxf(mx, __shfl_xor(mx, 4, 32));
            mx = fmaxf(mx, __shfl_xor(mx, 8, 32));
            const float mnew = fmaxf(mrow[v], mx);
            pscale[v] = __expf(mrow[v] - mnew);
            const float p0 = __expf(s[0][v] - mnew);
            const float p1 = __expf(s[1][v] - mnew);
            s[0][v] = p0; s[1][v] = p1;
            float rs = p0 + p1;
            rs += __shfl_xor(rs, 1, 32);
            rs += __shfl_xor(rs, 2, 32);
            rs += __shfl_xor(rs, 4, 32);
            rs += __shfl_xor(rs, 8, 32);
            lrow[v] = lrow[v] * pscale[v] + rs;
            mrow[v] = mnew;
        }
#pragma unroll
        for (int nt = 0; nt < 4; ++nt)
#pragma unroll
            for (int v = 0; v < 8; ++v) o[nt][v] *= pscale[v];

        // C-layout P -> LDS row-major [16][32] -> A-layout (in-order DS, same wave)
        __bf16* pl = lds[wave];
#pragma unroll
        for (int t = 0; t < 2; ++t)
#pragma unroll
            for (int v = 0; v < 8; ++v)
                pl[(8 * half + v) * 32 + t * 16 + nlo] = (__bf16)s[t][v];

        TileVec pa;
        {
            const int r = nlo, ks = half * 8;
            pa.h.lo = *(const v8bf*)&pl[r * 32 + ks];
            pa.h.hi = *(const v8bf*)&pl[r * 32 + 16 + ks];
        }
#pragma unroll
        for (int nt = 0; nt < 4; ++nt) {
            v16bf vb = load_tile16(vbase, nt * 16, S_LEN, kc0);
            o[nt] = wmma_bf16(pa.full, vb, o[nt]);
        }
    }

#pragma unroll
    for (int v = 0; v < 8; ++v) lrow[v] = 1.0f / lrow[v];
    __bf16* obase = ob + ((size_t)b * S_LEN) * D_DIM + (size_t)h * HD;
#pragma unroll
    for (int nt = 0; nt < 4; ++nt) {
        const int n = nt * 16 + nlo;
#pragma unroll
        for (int v = 0; v < 8; ++v) {
            const int sr = q0 + 8 * half + v;
            obase[(size_t)sr * D_DIM + n] = (__bf16)(o[nt][v] * lrow[v]);
        }
    }
}

// ---------------------------------------------------------------- launch ---
extern "C" void kernel_launch(void* const* d_in, const int* in_sizes, int n_in,
                              void* d_out, int out_size, void* d_ws, size_t ws_size,
                              hipStream_t stream) {
    const float* x    = (const float*)d_in[0];
    const float* mask = (const float*)d_in[1];
    const float* wq   = (const float*)d_in[2];
    const float* bq   = (const float*)d_in[3];
    const float* wk   = (const float*)d_in[4];
    const float* bk   = (const float*)d_in[5];
    const float* wv   = (const float*)d_in[6];
    const float* bv   = (const float*)d_in[7];
    const float* wo   = (const float*)d_in[8];
    const float* bo   = (const float*)d_in[9];

    char* ws = (char*)d_ws;
    const size_t MB = 1u << 20;
    __bf16* xb  = (__bf16*)(ws + 0 * MB);    // 8 MiB  [4096,1024]
    __bf16* wqb = (__bf16*)(ws + 8 * MB);    // 2 MiB each
    __bf16* wkb = (__bf16*)(ws + 10 * MB);
    __bf16* wvb = (__bf16*)(ws + 12 * MB);
    __bf16* wob = (__bf16*)(ws + 14 * MB);
    __bf16* qbb = (__bf16*)(ws + 16 * MB);   // 8 MiB
    __bf16* kbb = (__bf16*)(ws + 24 * MB);   // 8 MiB
    __bf16* vtb = (__bf16*)(ws + 32 * MB);   // 8 MiB  [B,H,Hd,S]
    __bf16* obb = (__bf16*)(ws + 40 * MB);   // 8 MiB  (total 48 MiB)

    const int nX = M_TOT * D_DIM;
    const int nW = D_DIM * D_DIM;
    cvt_f32_bf16<<<(nX + 255) / 256, 256, 0, stream>>>(x,  xb,  nX);
    cvt_f32_bf16<<<(nW + 255) / 256, 256, 0, stream>>>(wq, wqb, nW);
    cvt_f32_bf16<<<(nW + 255) / 256, 256, 0, stream>>>(wk, wkb, nW);
    cvt_f32_bf16<<<(nW + 255) / 256, 256, 0, stream>>>(wv, wvb, nW);
    cvt_f32_bf16<<<(nW + 255) / 256, 256, 0, stream>>>(wo, wob, nW);

    dim3 ggrid(M_TOT / 128, D_DIM / 128);    // 32 x 8
    gemm_xw<<<ggrid, 256, 0, stream>>>(xb, wqb, bq, qbb, 0);
    gemm_xw<<<ggrid, 256, 0, stream>>>(xb, wkb, bk, kbb, 0);
    gemm_xw<<<ggrid, 256, 0, stream>>>(xb, wvb, bv, vtb, 1);

    attn_flash<<<512, 256, 0, stream>>>(qbb, kbb, vtb, mask, obb);

    gemm_xw<<<ggrid, 256, 0, stream>>>(obb, wob, bo, d_out, 2);
}
